// Decoder_34282428957258
// MI455X (gfx1250) — matrix operand order repaired and checked by reading, not compiled, not run on hardware
//
#include <hip/hip_runtime.h>
#include <hip/hip_bf16.h>

// ---------------------------------------------------------------------------
// Types for CDNA5 WMMA
// ---------------------------------------------------------------------------
typedef __attribute__((ext_vector_type(16))) __bf16 v16bf;
typedef __attribute__((ext_vector_type(8)))  __bf16 v8bf;
typedef __attribute__((ext_vector_type(2)))  __bf16 v2bf;
typedef __attribute__((ext_vector_type(8)))  float  v8f;

#define B_  128
#define S_  128
#define H_  1024
#define E_  512
#define V_  32000

__device__ __forceinline__ unsigned short f2bf(float f) {
    unsigned int u = __float_as_uint(f);
    u += 0x7fffu + ((u >> 16) & 1u);          // round-to-nearest-even
    return (unsigned short)(u >> 16);
}

__device__ __forceinline__ unsigned int pack2(float a, float b) {
#if defined(__has_builtin)
#if __has_builtin(__builtin_amdgcn_cvt_pk_bf16_f32)
    union { v2bf v; unsigned int u; } cv;
    cv.v = __builtin_amdgcn_cvt_pk_bf16_f32(a, b);   // low = a, high = b
    return cv.u;
#else
    return (unsigned int)f2bf(a) | ((unsigned int)f2bf(b) << 16);
#endif
#else
    return (unsigned int)f2bf(a) | ((unsigned int)f2bf(b) << 16);
#endif
}

__device__ __forceinline__ v8f zero8() {
    v8f z;
#pragma unroll
    for (int i = 0; i < 8; ++i) z[i] = 0.0f;
    return z;
}

__device__ __forceinline__ v16bf cat8(v8bf lo, v8bf hi) {
    return __builtin_shufflevector(lo, hi, 0,1,2,3,4,5,6,7,8,9,10,11,12,13,14,15);
}

// A fragment: 16x32 bf16, lane l(0..15)=row l with K {h*8..h*8+7, 16+h*8..16+h*8+7}
__device__ __forceinline__ v16bf frag_a(const unsigned short* sA, int rowBase, int lane) {
    const int l = lane & 15, h = lane >> 4;
    const unsigned short* p = sA + (rowBase + l) * 32;
    v8bf lo = *reinterpret_cast<const v8bf*>(p + h * 8);
    v8bf hi = *reinterpret_cast<const v8bf*>(p + 16 + h * 8);
    return cat8(lo, hi);
}

// B fragment: 32x16 bf16, lane holds one column; sB stored as [n][32 k] (weight rows)
__device__ __forceinline__ v16bf frag_b(const unsigned short* sB, int colBase, int lane) {
    const int l = lane & 15, h = lane >> 4;
    const unsigned short* p = sB + (colBase + l) * 32;
    v8bf lo = *reinterpret_cast<const v8bf*>(p + h * 8);
    v8bf hi = *reinterpret_cast<const v8bf*>(p + 16 + h * 8);
    return cat8(lo, hi);
}

__device__ __forceinline__ v8f wmma_bf16(v16bf a, v16bf b, v8f c) {
    return __builtin_amdgcn_wmma_f32_16x16x32_bf16(false, a, false, b, (short)0, c,
                                                   false, false);
}

// ---------------------------------------------------------------------------
// Kernel 1: embedding gather + bf16 staging of decoder_hidden
// ---------------------------------------------------------------------------
__global__ __launch_bounds__(256) void k_prep(const int* __restrict__ tl,
                                              const float* __restrict__ dec_h,
                                              const float* __restrict__ emb,
                                              unsigned int* __restrict__ x_bf32,
                                              unsigned int* __restrict__ c1_bf32,
                                              unsigned int* __restrict__ hid_bf32) {
    int p = blockIdx.x * 256 + threadIdx.x;
    if (p < B_ * E_ / 2) {
        int b = p / 256, e2 = p % 256;                // 256 float2 per row of 512
        const float2 v = *reinterpret_cast<const float2*>(
            &emb[(size_t)tl[b] * E_ + e2 * 2]);
        unsigned int bf = pack2(v.x, v.y);
        x_bf32[b * 1280 + e2] = bf;                   // x row = 2560 bf16 = 1280 u32
        c1_bf32[b * 1792 + 1536 + e2] = bf;           // c1 row = 3584 bf16 = 1792 u32
    } else {
        int q = p - B_ * E_ / 2;
        if (q < B_ * H_ / 2) {
            int b = q >> 9, j2 = q & 511;             // 512 float2 per row of 1024
            const float2 v = *reinterpret_cast<const float2*>(
                &dec_h[(size_t)b * H_ + j2 * 2]);
            hid_bf32[b * 512 + j2] = pack2(v.x, v.y);
        }
    }
}

// ---------------------------------------------------------------------------
// Kernel 2: fused attention energy GEMM + tanh + W2 dot (partial scores)
//   grid = (128 M-blocks of 128 rows, 16 N-chunks of 64), block = 256
//   M-block = one batch row b = blockIdx.x; A row r corresponds to s = r:
//   A row = [enc[s][b][0:2048), hid[b][0:1024)]
// ---------------------------------------------------------------------------
__global__ __launch_bounds__(256) void k_energy(const float* __restrict__ enc,
                                                const float* __restrict__ dec_h,
                                                const float* __restrict__ W1,
                                                const float* __restrict__ b1,
                                                const float* __restrict__ W2,
                                                float* __restrict__ scores_part) {
    __shared__ __align__(16) unsigned int sA32[128 * 16];   // 128 x 32 bf16
    __shared__ __align__(16) unsigned int sB32[64 * 16];    // 64 x 32 bf16
    __shared__ float cbuf[128 * 64];
    const unsigned short* sA = (const unsigned short*)sA32;
    const unsigned short* sB = (const unsigned short*)sB32;

    const int tid  = threadIdx.x;
    const int lane = tid & 31, wave = tid >> 5;
    const int waveM = wave >> 1, waveN = wave & 1;
    const int bidx = blockIdx.x;                 // batch row for this M-block
    const int nb = blockIdx.y * 64;

    v8f acc[2][2];
#pragma unroll
    for (int mi = 0; mi < 2; ++mi)
#pragma unroll
        for (int ni = 0; ni < 2; ++ni) acc[mi][ni] = zero8();

    for (int k0 = 0; k0 < 3072; k0 += 32) {
        __syncthreads();
        // ---- A tile (128 rows x 32 k), phase selected per tile ----
        if (k0 < 2048) {
#pragma unroll
            for (int i = 0; i < 4; ++i) {
                int e4 = i * 256 + tid;               // 0..1023 float4s
                int r = e4 >> 3, c4 = e4 & 7;         // s = r
                const float4 v = *reinterpret_cast<const float4*>(
                    &enc[((size_t)(r * B_ + bidx)) * 2048 + k0 + c4 * 4]);
                sA32[r * 16 + c4 * 2]     = pack2(v.x, v.y);
                sA32[r * 16 + c4 * 2 + 1] = pack2(v.z, v.w);
            }
        } else {
            // broadcast hidden: every row identical
#pragma unroll
            for (int i = 0; i < 4; ++i) {
                int e4 = i * 256 + tid;
                int r = e4 >> 3, c4 = e4 & 7;
                const float4 v = *reinterpret_cast<const float4*>(
                    &dec_h[(size_t)bidx * H_ + (k0 - 2048) + c4 * 4]);
                sA32[r * 16 + c4 * 2]     = pack2(v.x, v.y);
                sA32[r * 16 + c4 * 2 + 1] = pack2(v.z, v.w);
            }
        }
        // ---- B tile: 64 W1 rows x 32 k ----
#pragma unroll
        for (int i = 0; i < 2; ++i) {
            int e4 = i * 256 + tid;                   // 0..511 float4s
            int r = e4 >> 3, c4 = e4 & 7;
            const float4 v = *reinterpret_cast<const float4*>(
                &W1[(size_t)(nb + r) * 3072 + k0 + c4 * 4]);
            sB32[r * 16 + c4 * 2]     = pack2(v.x, v.y);
            sB32[r * 16 + c4 * 2 + 1] = pack2(v.z, v.w);
        }
        if (k0 + 32 < 3072 && tid < 64)
            __builtin_prefetch(&W1[(size_t)(nb + tid) * 3072 + k0 + 32], 0, 1);
        __syncthreads();

#pragma unroll
        for (int mi = 0; mi < 2; ++mi) {
            v16bf af = frag_a(sA, waveM * 32 + mi * 16, lane);
#pragma unroll
            for (int ni = 0; ni < 2; ++ni) {
                v16bf bf = frag_b(sB, waveN * 32 + ni * 16, lane);
                acc[mi][ni] = wmma_bf16(af, bf, acc[mi][ni]);
            }
        }
    }

    // epilogue: cbuf[m][n] = tanh(acc + b1[n]) * W2[n]
    const int l = lane & 15, h = lane >> 4;
    float b1v[2], w2v[2];
#pragma unroll
    for (int ni = 0; ni < 2; ++ni) {
        int n = nb + waveN * 32 + ni * 16 + l;
        b1v[ni] = b1[n];
        w2v[ni] = W2[n];
    }
    __syncthreads();
#pragma unroll
    for (int mi = 0; mi < 2; ++mi)
#pragma unroll
        for (int ni = 0; ni < 2; ++ni) {
            int n_local = waveN * 32 + ni * 16 + l;
#pragma unroll
            for (int i = 0; i < 8; ++i) {
                int m_local = waveM * 32 + mi * 16 + i + 8 * h;
                cbuf[m_local * 64 + n_local] =
                    tanhf(acc[mi][ni][i] + b1v[ni]) * w2v[ni];
            }
        }
    __syncthreads();
    if (tid < 128) {
        float s = 0.0f;
#pragma unroll
        for (int j = 0; j < 64; ++j) {
            int jj = (j + tid) & 63;                  // rotate to avoid bank conflicts
            s += cbuf[tid * 64 + jj];
        }
        scores_part[(size_t)blockIdx.y * (B_ * S_) + bidx * S_ + tid] = s;
    }
}

// ---------------------------------------------------------------------------
// Kernel 3: softmax over S per batch row + context vector
// ---------------------------------------------------------------------------
__global__ __launch_bounds__(256) void k_softmax_ctx(const float* __restrict__ scores_part,
                                                     const float* __restrict__ b2,
                                                     const float* __restrict__ enc,
                                                     unsigned short* __restrict__ x_bf,
                                                     unsigned short* __restrict__ c1_bf) {
    __shared__ float alpha[128];
    __shared__ float red[256];
    const int b = blockIdx.x, tid = threadIdx.x;

    if (tid < 128) {
        float v = b2[0];
#pragma unroll
        for (int c = 0; c < 16; ++c)
            v += scores_part[(size_t)c * (B_ * S_) + b * S_ + tid];
        alpha[tid] = v;
    }
    __syncthreads();
    red[tid] = (tid < 128) ? alpha[tid] : -3.4e38f;
    __syncthreads();
    for (int off = 128; off > 0; off >>= 1) {
        if (tid < off) red[tid] = fmaxf(red[tid], red[tid + off]);
        __syncthreads();
    }
    float mx = red[0];
    __syncthreads();
    if (tid < 128) {
        float e = expf(alpha[tid] - mx);
        alpha[tid] = e;
        red[tid] = e;
    } else red[tid] = 0.0f;
    __syncthreads();
    for (int off = 128; off > 0; off >>= 1) {
        if (tid < off) red[tid] += red[tid + off];
        __syncthreads();
    }
    float inv = 1.0f / red[0];
    __syncthreads();

    for (int d = tid; d < 2048; d += 256) {
        float acc = 0.0f;
#pragma unroll 4
        for (int s = 0; s < S_; ++s)
            acc += alpha[s] * enc[((size_t)(s * B_ + b)) * 2048 + d];
        acc *= inv;
        unsigned short bf = f2bf(acc);
        x_bf[b * 2560 + 512 + d] = bf;
        c1_bf[b * 3584 + 1024 + d] = bf;
    }
}

// ---------------------------------------------------------------------------
// Kernel 4 (generic): C[128 x N] = A_bf16[128 x K] * W[N x K]^T + bias
//   block tile 128(M) x 128(N); 8 waves in 4(M) x 2(N); wave tile 32 x 64
//   requires: lda % 8 == 0, ldw % 4 == 0, K % 32 == 0, N % 128 == 0
// ---------------------------------------------------------------------------
__global__ __launch_bounds__(256) void k_gemm(const unsigned short* __restrict__ A,
                                              int lda,
                                              const float* __restrict__ W,
                                              int ldw,
                                              const float* __restrict__ bias,
                                              float* __restrict__ C,
                                              int ldc,
                                              int K) {
    __shared__ __align__(16) unsigned int sA32[128 * 16];   // 128 x 32 bf16
    __shared__ __align__(16) unsigned int sB32[128 * 16];   // 128 x 32 bf16
    const unsigned short* sA = (const unsigned short*)sA32;
    const unsigned short* sB = (const unsigned short*)sB32;

    const int tid  = threadIdx.x;
    const int lane = tid & 31, wave = tid >> 5;
    const int waveM = wave >> 1, waveN = wave & 1;          // 4 x 2
    const int nb = blockIdx.x * 128;

    v8f acc[2][4];
#pragma unroll
    for (int mi = 0; mi < 2; ++mi)
#pragma unroll
        for (int ni = 0; ni < 4; ++ni) acc[mi][ni] = zero8();

    for (int k0 = 0; k0 < K; k0 += 32) {
        __syncthreads();
        // A tile: straight uint4 copy of bf16 rows (2 x 16B per thread)
#pragma unroll
        for (int i = 0; i < 2; ++i) {
            int e8 = i * 256 + tid;                   // 0..511 groups of 8 bf16
            int r = e8 >> 2, c8 = e8 & 3;
            const uint4 v = *reinterpret_cast<const uint4*>(
                &A[(size_t)r * lda + k0 + c8 * 8]);
            *reinterpret_cast<uint4*>(&sA32[r * 16 + c8 * 4]) = v;
        }
        // B tile: 128 W rows x 32 k (4 float4 per thread), convert to bf16
#pragma unroll
        for (int i = 0; i < 4; ++i) {
            int e4 = i * 256 + tid;                   // 0..1023 float4s
            int r = e4 >> 3, c4 = e4 & 7;
            const float4 v = *reinterpret_cast<const float4*>(
                &W[(size_t)(nb + r) * ldw + k0 + c4 * 4]);
            sB32[r * 16 + c4 * 2]     = pack2(v.x, v.y);
            sB32[r * 16 + c4 * 2 + 1] = pack2(v.z, v.w);
        }
        if (k0 + 32 < K && tid < 128)
            __builtin_prefetch(&W[(size_t)(nb + tid) * ldw + k0 + 32], 0, 1);
        __syncthreads();

        v16bf af[2], bfr[4];
#pragma unroll
        for (int mi = 0; mi < 2; ++mi)
            af[mi] = frag_a(sA, waveM * 32 + mi * 16, lane);
#pragma unroll
        for (int ni = 0; ni < 4; ++ni)
            bfr[ni] = frag_b(sB, waveN * 64 + ni * 16, lane);
#pragma unroll
        for (int mi = 0; mi < 2; ++mi)
#pragma unroll
            for (int ni = 0; ni < 4; ++ni)
                acc[mi][ni] = wmma_bf16(af[mi], bfr[ni], acc[mi][ni]);
    }

    const int l = lane & 15, h = lane >> 4;
#pragma unroll
    for (int ni = 0; ni < 4; ++ni) {
        int n = nb + waveN * 64 + ni * 16 + l;
        float bv = bias[n];
#pragma unroll
        for (int mi = 0; mi < 2; ++mi)
#pragma unroll
            for (int i = 0; i < 8; ++i) {
                int m = waveM * 32 + mi * 16 + i + 8 * h;
                C[(size_t)m * ldc + n] = acc[mi][ni][i] + bv;
            }
    }
}

// ---------------------------------------------------------------------------
// Kernel 5: GRU pointwise combine; writes h_new (fp32 output) + c1_bf[:,0:1024)
// ---------------------------------------------------------------------------
__global__ __launch_bounds__(256) void k_gru(const float* __restrict__ gx,
                                             const float* __restrict__ gh,
                                             const float* __restrict__ dec_h,
                                             float* __restrict__ h_out,
                                             unsigned short* __restrict__ c1_bf) {
    int idx = blockIdx.x * 256 + threadIdx.x;          // B*H = 131072
    int b = idx >> 10, j = idx & 1023;
    const float* gxr = gx + (size_t)b * 3072;
    const float* ghr = gh + (size_t)b * 3072;
    float xr = gxr[j],        hr = ghr[j];
    float xz = gxr[1024 + j], hz = ghr[1024 + j];
    float xn = gxr[2048 + j], hn = ghr[2048 + j];
    float r = 1.0f / (1.0f + expf(-(xr + hr)));
    float z = 1.0f / (1.0f + expf(-(xz + hz)));
    float n = tanhf(xn + r * hn);
    float h = (1.0f - z) * n + z * dec_h[b * H_ + j];
    h_out[idx] = h;
    c1_bf[b * 3584 + j] = f2bf(h);
}

// ---------------------------------------------------------------------------
// Host launcher
// ---------------------------------------------------------------------------
extern "C" void kernel_launch(void* const* d_in, const int* in_sizes, int n_in,
                              void* d_out, int out_size, void* d_ws, size_t ws_size,
                              hipStream_t stream) {
    (void)in_sizes; (void)n_in; (void)out_size; (void)ws_size;

    const int*   tl    = (const int*)  d_in[0];
    const float* dec_h = (const float*)d_in[1];
    const float* enc   = (const float*)d_in[2];
    const float* emb   = (const float*)d_in[3];
    const float* W1    = (const float*)d_in[4];
    const float* b1    = (const float*)d_in[5];
    const float* W2    = (const float*)d_in[6];
    const float* b2    = (const float*)d_in[7];
    const float* Wih   = (const float*)d_in[8];
    const float* Whh   = (const float*)d_in[9];
    const float* bih   = (const float*)d_in[10];
    const float* bhh   = (const float*)d_in[11];
    const float* Wout  = (const float*)d_in[12];
    const float* bout  = (const float*)d_in[13];

    float* pred  = (float*)d_out;
    float* h_out = pred + (size_t)B_ * V_;

    char* ws = (char*)d_ws;
    float*          scores_part = (float*)ws;                        // 16*16384*4 = 1048576
    unsigned short* x_bf        = (unsigned short*)(ws + 1048576);   // 128*2560*2 = 655360
    unsigned short* c1_bf       = (unsigned short*)(ws + 1703936);   // 128*3584*2 = 917504
    unsigned short* hid_bf      = (unsigned short*)(ws + 2621440);   // 128*1024*2 = 262144
    float*          gx          = (float*)(ws + 2883584);            // 128*3072*4 = 1572864
    float*          gh          = (float*)(ws + 4456448);            // 128*3072*4 = 1572864
                                                                     // total 6029312 B

    // 1. embedding gather + bf16 staging (float2 granularity)
    k_prep<<<dim3((B_ * E_ / 2 + B_ * H_ / 2 + 255) / 256), dim3(256), 0, stream>>>(
        tl, dec_h, emb, (unsigned int*)x_bf, (unsigned int*)c1_bf,
        (unsigned int*)hid_bf);

    // 2. attention energy + fused tanh/W2 partial scores
    k_energy<<<dim3(B_, 16), dim3(256), 0, stream>>>(
        enc, dec_h, W1, b1, W2, scores_part);

    // 3. softmax + context (fills a into x_bf and c1_bf)
    k_softmax_ctx<<<dim3(B_), dim3(256), 0, stream>>>(
        scores_part, b2, enc, x_bf, c1_bf);

    // 4a. gx = x @ Wih^T + bih   (M=128, K=2560, N=3072)
    k_gemm<<<dim3(3072 / 128), dim3(256), 0, stream>>>(
        x_bf, 2560, Wih, 2560, bih, gx, 3072, 2560);

    // 4b. gh = h @ Whh^T + bhh   (M=128, K=1024, N=3072)
    k_gemm<<<dim3(3072 / 128), dim3(256), 0, stream>>>(
        hid_bf, 1024, Whh, 1024, bhh, gh, 3072, 1024);

    // 5. GRU combine -> h_new (output) + c1_bf[:,0:1024)
    k_gru<<<dim3(B_ * H_ / 256), dim3(256), 0, stream>>>(
        gx, gh, dec_h, h_out, c1_bf);

    // 6. pred = c1 @ Wout^T + bout  (M=128, K=3584, N=32000)
    k_gemm<<<dim3(V_ / 128), dim3(256), 0, stream>>>(
        c1_bf, 3584, Wout, 3584, bout, pred, V_, 3584);
}
